// Attention_gumbel_51874615001684
// MI455X (gfx1250) — matrix operand order repaired and checked by reading, not compiled, not run on hardware
//
#include <hip/hip_runtime.h>
#include <hip/hip_bf16.h>
#include <stdint.h>

// ---------------------------------------------------------------------------
// Attention + gumbel-mask forward for MI455X (gfx1250, wave32, WMMA).
// B=4 N=1024 C=384 H=6 D=64.  Outputs (concat): out(4,1024,384),
// attn_mean(4,6,1024,1024), u(4,6,1024,1024) -- 208MB f32 => HBM-write bound.
// GEMMs: v_wmma_f32_16x16x32_f16; LDS tiles filled with
// global_load_async_to_lds_b128 (ASYNCcnt), DOUBLE-BUFFERED so the DMA of
// tile i+1 overlaps the WMMAs of tile i (wait asynccnt<=2 = in-order done).
// All B-operands pre-transposed (weights at cvt; V written transposed by the
// QKV epilogue) so every tile copy is one contiguous 16B async op per lane.
// ---------------------------------------------------------------------------

#define B_   4
#define N_   1024
#define C_   384
#define H_   6
#define D_   64
#define CK_  1152          // 3*C
#define BN_  4096          // B*N
#define M_   1024
#define SCALE_ 0.125f      // D^-0.5
#define HALF_IDX_ 12582912 // (B*H*N*M)/2

typedef __attribute__((ext_vector_type(8)))  _Float16 v8h;
typedef __attribute__((ext_vector_type(16))) _Float16 v16h;
typedef __attribute__((ext_vector_type(8)))  float    v8f;

// ----------------------- async global->LDS copy (CDNA5) ---------------------
// VDST VGPR = LDS byte address (ISA 10.2: LDS addr = flat addr[31:0]);
// 64-bit global address in VGPR pair; tracked by ASYNCcnt (in-order done).
static __device__ __forceinline__ void async_copy_b128(void* lds_dst, const void* gsrc) {
  uint32_t lds_off = (uint32_t)(uintptr_t)lds_dst;
  asm volatile("global_load_async_to_lds_b128 %0, %1, off"
               :: "v"(lds_off), "v"(gsrc) : "memory");
}
static __device__ __forceinline__ void wait_async0() {
  asm volatile("s_wait_asynccnt 0x0" ::: "memory");
}
static __device__ __forceinline__ void wait_async2() {   // allow 2 in flight
  asm volatile("s_wait_asynccnt 0x2" ::: "memory");
}

// ----------------------- WMMA fragment helpers -----------------------------
static __device__ __forceinline__ v16h frag16(v8h lo, v8h hi) {
  v16h r;
#pragma unroll
  for (int j = 0; j < 8; ++j) { r[j] = lo[j]; r[j + 8] = hi[j]; }
  return r;
}

// A fragment (16x32 f16, row-major source, lda halves/row).
// lane m=l%16 g=l/16: halves j = K 16g + j, j=0..15 (contiguous, matches B)
static __device__ __forceinline__ v16h load_frag_a(const _Float16* base, int lda, int lane) {
  const int m = lane & 15, g = lane >> 4;
  const _Float16* p = base + m * lda + 16 * g;
  return frag16(*(const v8h*)p, *(const v8h*)(p + 8));
}

// B fragment (32x16) from an n-major (transposed) tile Bt[n][k], ldb halves/row.
// lane n=l%16, g=l/16: halves j = Bt[n][16g + j], j=0..15 (contiguous 32B)
static __device__ __forceinline__ v16h load_frag_bt(const _Float16* base, int ldb, int lane) {
  const int n = lane & 15, g = lane >> 4;
  const _Float16* p = base + n * ldb + 16 * g;
  return frag16(*(const v8h*)p, *(const v8h*)(p + 8));
}

static __device__ __forceinline__ v8f wmma32(v16h a, v16h b, v8f c) {
  return __builtin_amdgcn_wmma_f32_16x16x32_f16(false, a, false, b, (short)0, c, false, false);
}

// ----------------------- threefry2x32 (JAX RNG) ----------------------------
__host__ __device__ static inline uint32_t rotl32(uint32_t v, int s) {
  return (v << s) | (v >> (32 - s));
}
__host__ __device__ static inline void threefry2x32(uint32_t k0, uint32_t k1,
                                                    uint32_t x0, uint32_t x1,
                                                    uint32_t& o0, uint32_t& o1) {
  const uint32_t k2 = k0 ^ k1 ^ 0x1BD11BDAu;
  x0 += k0; x1 += k1;
#define TF_R(r) { x0 += x1; x1 = rotl32(x1, r); x1 ^= x0; }
  TF_R(13) TF_R(15) TF_R(26) TF_R(6)   x0 += k1; x1 += k2 + 1u;
  TF_R(17) TF_R(29) TF_R(16) TF_R(24)  x0 += k2; x1 += k0 + 2u;
  TF_R(13) TF_R(15) TF_R(26) TF_R(6)   x0 += k0; x1 += k1 + 3u;
  TF_R(17) TF_R(29) TF_R(16) TF_R(24)  x0 += k1; x1 += k2 + 4u;
  TF_R(13) TF_R(15) TF_R(26) TF_R(6)   x0 += k2; x1 += k0 + 5u;
#undef TF_R
  o0 = x0; o1 = x1;
}

__device__ static __forceinline__ float bits_to_gumbel(uint32_t r) {
  const float TINY = 1.1754944e-38f;
  float f = __uint_as_float(0x3f800000u | (r >> 9)) - 1.0f;   // [0,1)
  float u = fmaxf(f * (1.0f - TINY) + TINY, TINY);            // jax uniform(tiny,1)
  return -__logf(-__logf(u));
}

// ----------------------- kernel 0: f32 -> f16 (+weight transpose) -----------
#define NX_  1572864
#define NWQ_ 442368
#define NWP_ 147456
__global__ __launch_bounds__(256) void cvt_kernel(
    const float* __restrict__ x, const float* __restrict__ wq, const float* __restrict__ wp,
    _Float16* __restrict__ xh, _Float16* __restrict__ wqhT, _Float16* __restrict__ wphT) {
  int i = blockIdx.x * 256 + threadIdx.x;
  if (i < NX_) {
    xh[i] = (_Float16)x[i];
  } else if (i < NX_ + NWQ_) {
    int j = i - NX_;                       // wq is (C, CK) row-major
    int kk = j / CK_, n = j - kk * CK_;
    wqhT[(size_t)n * C_ + kk] = (_Float16)wq[j];   // (CK, C)
  } else if (i < NX_ + NWQ_ + NWP_) {
    int j = i - NX_ - NWQ_;                // wp is (C, C) row-major
    int kk = j / C_, n = j - kk * C_;
    wphT[(size_t)n * C_ + kk] = (_Float16)wp[j];   // transposed
  }
}

// ----------------------- kernel 1: qkv = x @ qkv_w + b ----------------------
// (4096x384)@(384x1152): 64x64 block tile, 8 waves of 16x32, K-step 32.
// Double-buffered async-DMA LDS tiles: DMA(i+1) overlaps WMMA(i).
__global__ __launch_bounds__(256) void qkv_gemm_kernel(
    const _Float16* __restrict__ xh, const _Float16* __restrict__ whT,
    const float* __restrict__ qkvb,
    _Float16* __restrict__ qh, _Float16* __restrict__ kh, _Float16* __restrict__ vT) {
  __shared__ __align__(16) _Float16 As[2][64 * 40];
  __shared__ __align__(16) _Float16 Bt[2][64 * 40];
  const int tid = threadIdx.x, lane = tid & 31, wave = tid >> 5;
  const int m0 = blockIdx.x * 64, n0 = blockIdx.y * 64;
  const int mi = wave & 3, ni = wave >> 2;
  const int r = tid >> 2, kc = (tid & 3) * 8;    // 64 rows x 32 halves per tile
  const _Float16* gA = &xh[(size_t)(m0 + r) * C_ + kc];
  const _Float16* gB = &whT[(size_t)(n0 + r) * C_ + kc];
  const int NSTEP = C_ / 32;                     // 12
  v8f acc[2] = {v8f{}, v8f{}};
  async_copy_b128(&As[0][r * 40 + kc], gA);
  async_copy_b128(&Bt[0][r * 40 + kc], gB);
  for (int i = 0; i < NSTEP; ++i) {
    const int cur = i & 1;
    if (i + 1 < NSTEP) {
      async_copy_b128(&As[1 - cur][r * 40 + kc], gA + 32 * (i + 1));
      async_copy_b128(&Bt[1 - cur][r * 40 + kc], gB + 32 * (i + 1));
      wait_async2();              // tile i complete (in-order), tile i+1 in flight
    } else {
      wait_async0();
    }
    __syncthreads();
    v16h a  = load_frag_a(&As[cur][mi * 16 * 40], 40, lane);
    v16h b0 = load_frag_bt(&Bt[cur][(ni * 32) * 40], 40, lane);
    v16h b1 = load_frag_bt(&Bt[cur][(ni * 32 + 16) * 40], 40, lane);
    acc[0] = wmma32(a, b0, acc[0]);
    acc[1] = wmma32(a, b1, acc[1]);
    __syncthreads();              // buffer reused at step i+2
  }
  const int g = lane >> 4, cl = lane & 15;
#pragma unroll
  for (int p = 0; p < 2; ++p) {
    int col = n0 + ni * 32 + p * 16 + cl;        // [0,1152)
    float bias = qkvb[col];
    int which = col / C_;                         // 0=q 1=k 2=v (no straddle: 384%16==0)
    int rem = col - which * C_;
    int h = rem >> 6, d = rem & 63;
#pragma unroll
    for (int j = 0; j < 8; ++j) {
      int row = m0 + mi * 16 + g * 8 + j;        // [0,4096)
      int b = row >> 10, n = row & 1023;
      _Float16 val = (_Float16)(acc[p][j] + bias);
      if (which == 0)      qh[((size_t)(b * H_ + h) * N_ + n) * D_ + d] = val;
      else if (which == 1) kh[((size_t)(b * H_ + h) * N_ + n) * D_ + d] = val;
      else                 vT[((size_t)(b * H_ + h) * D_ + d) * N_ + n] = val;  // transposed
    }
  }
}

// ----------------------- kernel 2: qk = q@k^T + row softmax stats -----------
// block = (b,h, 16-row n-tile); 8 waves x 8 m-tiles cover all 1024 keys.
// k is row-major (m,d) == Bt layout, so fragments load straight from global.
__global__ __launch_bounds__(256) void scores_kernel(
    const _Float16* __restrict__ qh, const _Float16* __restrict__ kh,
    float* __restrict__ qkraw, float* __restrict__ stats) {
  __shared__ float smax[8][16];
  __shared__ float ssum[8][16];
  const int tid = threadIdx.x, lane = tid & 31, wave = tid >> 5;
  const int bh = blockIdx.x;              // b*H+h
  const int n0 = blockIdx.y * 16;
  const int g = lane >> 4, cl = lane & 15;
  const _Float16* qbase = qh + ((size_t)bh * N_ + n0) * D_;
  v16h a0 = load_frag_a(qbase, 64, lane);        // d 0..31
  v16h a1 = load_frag_a(qbase + 32, 64, lane);   // d 32..63
  float sc[8][8];
#pragma unroll
  for (int i = 0; i < 8; ++i) {
    int mt = wave + 8 * i;
    const _Float16* kb = kh + ((size_t)bh * N_ + mt * 16) * D_;
    v16h b0 = load_frag_bt(kb, 64, lane);
    v16h b1 = load_frag_bt(kb + 32, 64, lane);
    v8f c = {};
    c = wmma32(a0, b0, c);
    c = wmma32(a1, b1, c);
    float* dst = qkraw + ((size_t)bh * N_ + n0 + g * 8) * M_ + mt * 16 + cl;
#pragma unroll
    for (int j = 0; j < 8; ++j) { dst[(size_t)j * M_] = c[j]; sc[i][j] = c[j]; }
  }
  // per-row (row = 8g+j) softmax stats over this wave's 128 keys, then x-wave
#pragma unroll
  for (int j = 0; j < 8; ++j) {
    float mx = sc[0][j];
#pragma unroll
    for (int i = 1; i < 8; ++i) mx = fmaxf(mx, sc[i][j]);
#pragma unroll
    for (int off = 1; off < 16; off <<= 1) mx = fmaxf(mx, __shfl_xor(mx, off, 32));
    mx *= SCALE_;                                   // stats on scaled scores
    float s = 0.0f;
#pragma unroll
    for (int i = 0; i < 8; ++i) s += __expf(sc[i][j] * SCALE_ - mx);
#pragma unroll
    for (int off = 1; off < 16; off <<= 1) s += __shfl_xor(s, off, 32);
    if (cl == 0) { smax[wave][g * 8 + j] = mx; ssum[wave][g * 8 + j] = s; }
  }
  __syncthreads();
  if (tid < 16) {
    float gm = smax[0][tid];
#pragma unroll
    for (int w = 1; w < 8; ++w) gm = fmaxf(gm, smax[w][tid]);
    float gs = 0.0f;
#pragma unroll
    for (int w = 0; w < 8; ++w) gs += ssum[w][tid] * __expf(smax[w][tid] - gm);
    size_t rr = (size_t)bh * N_ + n0 + tid;
    stats[2 * rr] = gm; stats[2 * rr + 1] = gs;
  }
}

// ----------------------- kernel 3: attn_mean, u, gumbel mask ----------------
// Each thread handles one (n,m) for batch pair {bb, bb+2}, all 6 channels.
// threefry counts (i, i+half) are exactly batches (b, b+2): one call, both used.
__global__ __launch_bounds__(256) void mask_kernel(
    const float* __restrict__ qkraw, const float* __restrict__ stats,
    const float* __restrict__ cwg, const float* __restrict__ cbg,
    float* __restrict__ out_attn, float* __restrict__ out_u,
    _Float16* __restrict__ attnh,
    uint32_t k0a, uint32_t k0b, uint32_t k1a, uint32_t k1b) {
  int linear = blockIdx.x * 256 + threadIdx.x;        // 2*1024*1024 threads
  int m = linear & (M_ - 1);
  int n = (linear >> 10) & (N_ - 1);
  int bb = linear >> 20;                              // 0 or 1
  float cw[H_][H_], cb[H_];
#pragma unroll
  for (int o = 0; o < H_; ++o) {
    cb[o] = cbg[o];
#pragma unroll
    for (int h = 0; h < H_; ++h) cw[o][h] = cwg[o * H_ + h];
  }
  float qv[2][H_], am[2][H_];
#pragma unroll
  for (int p = 0; p < 2; ++p) {
    int b = bb + 2 * p;
#pragma unroll
    for (int h = 0; h < H_; ++h) {
      size_t row = (size_t)(b * H_ + h) * N_ + n;
      size_t idx = row * M_ + m;
      float s = qkraw[idx];
      float mx = stats[2 * row], sm = stats[2 * row + 1];
      float a = __expf(s * SCALE_ - mx) / sm;
      qv[p][h] = s; am[p][h] = a;
      out_attn[idx] = a;
    }
  }
#pragma unroll
  for (int o = 0; o < H_; ++o) {
    float u0 = cb[o], u1 = cb[o];
#pragma unroll
    for (int h = 0; h < H_; ++h) { u0 += cw[o][h] * qv[0][h]; u1 += cw[o][h] * qv[1][h]; }
    u0 = 0.5f * (tanhf(u0) + 1.0f);                  // log(exp(u))==u
    u1 = 0.5f * (tanhf(u1) + 1.0f);
    size_t i0 = ((size_t)(bb * H_ + o) * N_ + n) * M_ + m;   // < HALF_IDX_
    size_t i1 = i0 + (size_t)HALF_IDX_;                      // batch bb+2
    out_u[i0] = u0; out_u[i1] = u1;
    uint32_t a0, a1, b0, b1;
    threefry2x32(k0a, k0b, (uint32_t)i0, (uint32_t)i1, a0, a1);  // g0 stream
    threefry2x32(k1a, k1b, (uint32_t)i0, (uint32_t)i1, b0, b1);  // g1 stream
    float g00 = bits_to_gumbel(a0), g01 = bits_to_gumbel(a1);
    float g10 = bits_to_gumbel(b0), g11 = bits_to_gumbel(b1);
    // forward value of straight-through mask == hard1
    float mk0 = ((1.0f - u0) + g10) > (u0 + g00) ? 1.0f : 0.0f;
    float mk1 = ((1.0f - u1) + g11) > (u1 + g01) ? 1.0f : 0.0f;
    attnh[i0] = (_Float16)(am[0][o] * mk0);
    attnh[i1] = (_Float16)(am[1][o] * mk1);
  }
}

// ----------------------- kernel 4: hid = attn @ v ---------------------------
// (per b,h) (1024x1024)@(1024x64) -> hidden (B,N, h*64+d) f16.
// V consumed from transposed vT(B,H,D,N): both tiles contiguous async DMA,
// double-buffered.
__global__ __launch_bounds__(256) void av_gemm_kernel(
    const _Float16* __restrict__ attnh, const _Float16* __restrict__ vT,
    _Float16* __restrict__ hid) {
  __shared__ __align__(16) _Float16 As[2][64 * 40];
  __shared__ __align__(16) _Float16 Vt[2][64 * 40];
  const int tid = threadIdx.x, lane = tid & 31, wave = tid >> 5;
  const int bh = blockIdx.x;             // 24
  const int n0 = blockIdx.y * 64;        // 16 tiles
  const int mi = wave & 3, ni = wave >> 2;
  const int r = tid >> 2, kc = (tid & 3) * 8;
  const _Float16* gA = &attnh[((size_t)bh * N_ + n0 + r) * M_ + kc];
  const _Float16* gB = &vT[((size_t)bh * D_ + r) * N_ + kc];
  const int NSTEP = M_ / 32;             // 32
  v8f acc[2] = {v8f{}, v8f{}};
  async_copy_b128(&As[0][r * 40 + kc], gA);
  async_copy_b128(&Vt[0][r * 40 + kc], gB);
  for (int i = 0; i < NSTEP; ++i) {
    const int cur = i & 1;
    if (i + 1 < NSTEP) {
      async_copy_b128(&As[1 - cur][r * 40 + kc], gA + 32 * (i + 1));
      async_copy_b128(&Vt[1 - cur][r * 40 + kc], gB + 32 * (i + 1));
      wait_async2();
    } else {
      wait_async0();
    }
    __syncthreads();
    v16h a  = load_frag_a(&As[cur][mi * 16 * 40], 40, lane);
    v16h b0 = load_frag_bt(&Vt[cur][(ni * 32) * 40], 40, lane);
    v16h b1 = load_frag_bt(&Vt[cur][(ni * 32 + 16) * 40], 40, lane);
    acc[0] = wmma32(a, b0, acc[0]);
    acc[1] = wmma32(a, b1, acc[1]);
    __syncthreads();
  }
  const int g = lane >> 4, cl = lane & 15;
  const int b = bh / H_, h = bh % H_;
#pragma unroll
  for (int p = 0; p < 2; ++p) {
    int d = ni * 32 + p * 16 + cl;
#pragma unroll
    for (int j = 0; j < 8; ++j) {
      int n = n0 + mi * 16 + g * 8 + j;
      hid[((size_t)(b * N_ + n)) * C_ + h * D_ + d] = (_Float16)acc[p][j];
    }
  }
}

// ----------------------- kernel 5: out = hid @ proj_w + b -------------------
__global__ __launch_bounds__(256) void proj_gemm_kernel(
    const _Float16* __restrict__ hid, const _Float16* __restrict__ whT,
    const float* __restrict__ pb, float* __restrict__ out) {
  __shared__ __align__(16) _Float16 As[2][64 * 40];
  __shared__ __align__(16) _Float16 Bt[2][64 * 40];
  const int tid = threadIdx.x, lane = tid & 31, wave = tid >> 5;
  const int m0 = blockIdx.x * 64, n0 = blockIdx.y * 64;
  const int mi = wave & 3, ni = wave >> 2;
  const int r = tid >> 2, kc = (tid & 3) * 8;
  const _Float16* gA = &hid[(size_t)(m0 + r) * C_ + kc];
  const _Float16* gB = &whT[(size_t)(n0 + r) * C_ + kc];
  const int NSTEP = C_ / 32;             // 12
  v8f acc[2] = {v8f{}, v8f{}};
  async_copy_b128(&As[0][r * 40 + kc], gA);
  async_copy_b128(&Bt[0][r * 40 + kc], gB);
  for (int i = 0; i < NSTEP; ++i) {
    const int cur = i & 1;
    if (i + 1 < NSTEP) {
      async_copy_b128(&As[1 - cur][r * 40 + kc], gA + 32 * (i + 1));
      async_copy_b128(&Bt[1 - cur][r * 40 + kc], gB + 32 * (i + 1));
      wait_async2();
    } else {
      wait_async0();
    }
    __syncthreads();
    v16h a  = load_frag_a(&As[cur][mi * 16 * 40], 40, lane);
    v16h b0 = load_frag_bt(&Bt[cur][(ni * 32) * 40], 40, lane);
    v16h b1 = load_frag_bt(&Bt[cur][(ni * 32 + 16) * 40], 40, lane);
    acc[0] = wmma32(a, b0, acc[0]);
    acc[1] = wmma32(a, b1, acc[1]);
    __syncthreads();
  }
  const int g = lane >> 4, cl = lane & 15;
#pragma unroll
  for (int p = 0; p < 2; ++p) {
    int col = n0 + ni * 32 + p * 16 + cl;
    float bias = pb[col];
#pragma unroll
    for (int j = 0; j < 8; ++j) {
      int row = m0 + mi * 16 + g * 8 + j;
      out[(size_t)row * C_ + col] = acc[p][j] + bias;
    }
  }
}

// ----------------------- host launcher --------------------------------------
extern "C" void kernel_launch(void* const* d_in, const int* in_sizes, int n_in,
                              void* d_out, int out_size, void* d_ws, size_t ws_size,
                              hipStream_t stream) {
  (void)in_sizes; (void)n_in; (void)out_size; (void)ws_size;
  const float* x      = (const float*)d_in[0];
  const float* qkv_w  = (const float*)d_in[1];
  const float* qkv_b  = (const float*)d_in[2];
  const float* proj_w = (const float*)d_in[3];
  const float* proj_b = (const float*)d_in[4];
  const float* conv_w = (const float*)d_in[5];
  const float* conv_b = (const float*)d_in[6];
  float* out = (float*)d_out;

  // workspace layout (bytes, all 256-aligned); total ~168.1 MB
  char* w = (char*)d_ws;
  _Float16* xh    = (_Float16*)(w + 0);          // (BN,C) f16
  _Float16* wqhT  = (_Float16*)(w + 3145728);    // (CK,C) f16 (transposed)
  _Float16* wphT  = (_Float16*)(w + 4030464);    // (C,C)  f16 (transposed)
  _Float16* qh    = (_Float16*)(w + 4325376);    // (B,H,N,D) f16
  _Float16* kh    = (_Float16*)(w + 7471104);    // (B,H,N,D) f16
  _Float16* vT    = (_Float16*)(w + 10616832);   // (B,H,D,N) f16 (transposed)
  _Float16* hid   = (_Float16*)(w + 13762560);   // (BN,C) f16
  _Float16* atth  = (_Float16*)(w + 16908288);   // (B,H,N,M) f16 masked attn
  float*    stats = (float*)   (w + 67239936);   // (B*H*N, 2) f32
  float*    qkf   = (float*)   (w + 67436544);   // (B,H,N,M) f32 raw scores

  float* out_proj = out;                 // 1,572,864
  float* out_attn = out + 1572864;       // 25,165,824
  float* out_u    = out + 26738688;      // 25,165,824

  // JAX: gk = key(42); fold_in(gk,0), fold_in(gk,1) -> threefry((0,42),(0,d))
  uint32_t ka0, ka1, kb0, kb1;
  threefry2x32(0u, 42u, 0u, 0u, ka0, ka1);
  threefry2x32(0u, 42u, 0u, 1u, kb0, kb1);

  cvt_kernel<<<8448, 256, 0, stream>>>(x, qkv_w, proj_w, xh, wqhT, wphT);
  qkv_gemm_kernel<<<dim3(64, 18), 256, 0, stream>>>(xh, wqhT, qkv_b, qh, kh, vT);
  scores_kernel<<<dim3(24, 64), 256, 0, stream>>>(qh, kh, qkf, stats);
  mask_kernel<<<8192, 256, 0, stream>>>(qkf, stats, conv_w, conv_b,
                                        out_attn, out_u, atth, ka0, ka1, kb0, kb1);
  av_gemm_kernel<<<dim3(24, 16), 256, 0, stream>>>(atth, vT, hid);
  proj_gemm_kernel<<<dim3(64, 6), 256, 0, stream>>>(hid, wphT, proj_b, out_proj);
}
